// Composition_Attention_67448166416733
// MI455X (gfx1250) — matrix operand order repaired and checked by reading, not compiled, not run on hardware
//
#include <hip/hip_runtime.h>
#include <hip/hip_bf16.h>

// ---------------------------------------------------------------------------
// Composition attention: per-row MLP score via bf16 WMMA, then deterministic
// segment softmax (batch ids are sorted).
//
// inputs (setup_inputs order):
//   0: x           [N,128] f32
//   1: batch       [N]     int (sorted, values in [0,B))
//   2: global_feat [B,103] f32
//   3: W1          [231,32] f32
//   4: b1          [32]     f32
//   5: W2          [32,1]   f32
//   6: b2          [1]      f32
// out: [N,1] f32
// ---------------------------------------------------------------------------

typedef __attribute__((ext_vector_type(16))) __bf16        v16bf;
typedef __attribute__((ext_vector_type(2)))  __bf16        v2bf;
typedef __attribute__((ext_vector_type(8)))  float         v8f;
typedef __attribute__((ext_vector_type(4)))  float         fx4;
typedef __attribute__((ext_vector_type(4)))  unsigned int  ux4;

#define NEURONS 128
#define GDIM    103
#define DIN     231    // 128 + 103
#define HID     32
#define ASTRIDE 264    // bf16 row stride in LDS A tile (256 + 8 pad: kills
                       // the 16-way bank conflict on fragment reads)

// pack two f32 into one dword of bf16 (v_cvt_pk_bf16_f32)
__device__ __forceinline__ unsigned int packbf(float lo, float hi) {
#if __has_builtin(__builtin_amdgcn_cvt_pk_bf16_f32)
  v2bf r = __builtin_amdgcn_cvt_pk_bf16_f32(lo, hi);
  return __builtin_bit_cast(unsigned int, r);
#else
  union { __bf16 h[2]; unsigned int u; } q;
  q.h[0] = (__bf16)lo;
  q.h[1] = (__bf16)hi;
  return q.u;
#endif
}

union Frag16 {            // 8 VGPRs = 16 bf16 per lane
  v16bf        v;
  ux4          q[2];
  unsigned int d[8];
};

__device__ __forceinline__ float softplusf(float v) {
  return fmaxf(v, 0.f) + log1pf(__expf(-fabsf(v)));
}

// ---------------------------------------------------------------------------
// Kernel 1: s[i] = W2 . softplus([x_i | gf_{batch[i]}] @ W1 + b1) + b2
// One wave per 16-row tile; K padded to 256 (8 chunks of 32); two N-tiles.
// A tile staged in per-wave LDS as bf16, fragments read with ds_load_b128.
// ---------------------------------------------------------------------------
__global__ __launch_bounds__(256) void mlp_score_kernel(
    const float* __restrict__ x, const int* __restrict__ batch,
    const float* __restrict__ gfeat, const float* __restrict__ W1,
    const float* __restrict__ b1, const float* __restrict__ W2,
    const float* __restrict__ b2, float* __restrict__ s_out, int n)
{
  // W1 packed in B-fragment order: [ntile][kc][i][lane] (dword = bf16 pair
  // (K=8*(lane/16)+2i+(i>=4?8:0), +1) for column ntile*16 + lane%16).
  __shared__ unsigned int sB[2 * 8 * 8 * 32];                       // 16 KB
  __shared__ __align__(16) unsigned short sA[8][16 * ASTRIDE];      // 66 KB
  __shared__ float sB1[HID];
  __shared__ float sW2[HID];
  __shared__ float sB2;

  for (int idx = threadIdx.x; idx < 2 * 8 * 8 * 32; idx += 256) {
    int lane = idx & 31;
    int i    = (idx >> 5) & 7;
    int kc   = (idx >> 8) & 7;
    int nt   = idx >> 11;
    int col  = nt * 16 + (lane & 15);
    int k0   = kc * 32 + (lane >> 4) * 8 + 2 * i + (i >= 4 ? 8 : 0);
    float f0 = (k0     < DIN) ? W1[(size_t)k0 * HID + col]       : 0.f;
    float f1 = (k0 + 1 < DIN) ? W1[(size_t)(k0 + 1) * HID + col] : 0.f;
    sB[idx] = packbf(f0, f1);
  }
  if (threadIdx.x < HID) {
    sB1[threadIdx.x] = b1[threadIdx.x];
    sW2[threadIdx.x] = W2[threadIdx.x];
  }
  if (threadIdx.x == 0) sB2 = b2[0];
  __syncthreads();

  const int lane  = threadIdx.x & 31;
  const int wave  = threadIdx.x >> 5;
  const int khalf = lane >> 4;    // 0: K 0..7/16..23 of chunk, 1: +8
  const int mrow  = lane & 15;    // A-matrix row within tile
  const int tiles = (n + 15) >> 4;

  unsigned short* const As = &sA[wave][0];

  for (int t = blockIdx.x * 8 + wave; t < tiles; t += gridDim.x * 8) {
    const int m0 = t * 16;
    // lane's own row id (lanes 16..31 mirror 0..15); broadcast below per row
    const int bid = batch[min(m0 + mrow, n - 1)];

    // ---- stage the 16 x 256 bf16 A tile into wave-private LDS ----------
#pragma unroll
    for (int r = 0; r < 16; ++r) {
      const int rowr = min(m0 + r, n - 1);
      // x region: cols 0..127, lane covers 4 consecutive, NT streamed
      const float* xr = x + (size_t)rowr * NEURONS;
      fx4 f = __builtin_nontemporal_load((const fx4*)(xr + 4 * lane));
      unsigned int d0 = packbf(f.x, f.y);
      unsigned int d1 = packbf(f.z, f.w);
      unsigned int* dst =
          (unsigned int*)(As + (size_t)r * ASTRIDE + 4 * lane);
      dst[0] = d0;
      dst[1] = d1;
      // gf region: cols 128..230 (+ zero pad to 255); 4B-aligned -> scalar
      const int bg = __shfl(bid, r, 32);
      const float* gr = gfeat + (size_t)bg * GDIM;
      float e0 = gr[2 * lane];            // 0..63  < 103, unconditional
      float e1 = gr[2 * lane + 1];
      int   j2 = 64 + 2 * lane;           // 64..126: guard tail
      float e2 = (j2     < GDIM) ? gr[j2]     : 0.f;
      float e3 = (j2 + 1 < GDIM) ? gr[j2 + 1] : 0.f;
      unsigned int* grow = (unsigned int*)(As + (size_t)r * ASTRIDE);
      grow[64 + lane] = packbf(e0, e1);   // cols 128+2L
      grow[96 + lane] = packbf(e2, e3);   // cols 192+2L (incl. zero pad)
    }

    // ---- 8 K-chunks x 2 N-tiles of WMMA --------------------------------
    v8f acc0 = {};   // cols 0..15
    v8f acc1 = {};   // cols 16..31

#pragma unroll
    for (int kc = 0; kc < 8; ++kc) {
      const int kb = kc * 32 + khalf * 8;   // first K of elems 0..7
      // A fragment: two contiguous 16B runs -> two ds_load_b128, no repack
      const unsigned short* arow = As + (size_t)mrow * ASTRIDE;
      Frag16 a;
      a.q[0] = *(const ux4*)(arow + kb);        // K kb   .. kb+7
      a.q[1] = *(const ux4*)(arow + kb + 16);   // K kb+16.. kb+23

      Frag16 bf0, bf1;
#pragma unroll
      for (int i = 0; i < 8; ++i) {
        bf0.d[i] = sB[(0 * 8 + kc) * 256 + i * 32 + lane];
        bf1.d[i] = sB[(1 * 8 + kc) * 256 + i * 32 + lane];
      }

      acc0 = __builtin_amdgcn_wmma_f32_16x16x32_bf16(
          false, a.v, false, bf0.v, (short)0, acc0, false, false);
      acc1 = __builtin_amdgcn_wmma_f32_16x16x32_bf16(
          false, a.v, false, bf1.v, (short)0, acc1, false, false);
    }

    // Epilogue. C/D layout: lane L, VGPR v -> (M = v + 8*(L/16), Ncol = L%16).
    const int col0 = mrow;
    const int col1 = 16 + mrow;
    float p[8];
#pragma unroll
    for (int v = 0; v < 8; ++v) {
      float h0 = softplusf(acc0[v] + sB1[col0]);
      float h1 = softplusf(acc1[v] + sB1[col1]);
      p[v] = h0 * sW2[col0] + h1 * sW2[col1];
    }
    // reduce across the 16 lanes of each half (masks 1,2,4,8 stay in-half)
#pragma unroll
    for (int v = 0; v < 8; ++v) {
      p[v] += __shfl_xor(p[v], 1, 32);
      p[v] += __shfl_xor(p[v], 2, 32);
      p[v] += __shfl_xor(p[v], 4, 32);
      p[v] += __shfl_xor(p[v], 8, 32);
    }
    if (mrow == 0) {
#pragma unroll
      for (int v = 0; v < 8; ++v) {
        int r = m0 + khalf * 8 + v;
        if (r < n) s_out[r] = p[v] + sB2;
      }
    }
  }
}

// ---------------------------------------------------------------------------
// Kernel 2: deterministic segment softmax over sorted batch ids.
// One thread per segment: binary-search range, then max / exp-sum / scale.
// ---------------------------------------------------------------------------
__device__ __forceinline__ int lower_bound_i(const int* __restrict__ a, int n,
                                             int key) {
  int lo = 0, hi = n;
  while (lo < hi) {
    int mid = (lo + hi) >> 1;
    if (a[mid] < key) lo = mid + 1; else hi = mid;
  }
  return lo;
}

__global__ __launch_bounds__(256) void segment_softmax_kernel(
    const float* __restrict__ s, const int* __restrict__ batch,
    float* __restrict__ out, int n, int nseg)
{
  int b = blockIdx.x * blockDim.x + threadIdx.x;
  if (b >= nseg) return;
  int lo = lower_bound_i(batch, n, b);
  int hi = lower_bound_i(batch, n, b + 1);
  float m = -__builtin_inff();
  for (int i = lo; i < hi; ++i) m = fmaxf(m, s[i]);
  float z = 0.f;
  for (int i = lo; i < hi; ++i) {
    float e = __expf(s[i] - m);
    out[i] = e;
    z += e;
  }
  float inv = 1.f / (z + 1e-16f);
  for (int i = lo; i < hi; ++i) out[i] *= inv;
}

// ---------------------------------------------------------------------------
extern "C" void kernel_launch(void* const* d_in, const int* in_sizes, int n_in,
                              void* d_out, int out_size, void* d_ws,
                              size_t ws_size, hipStream_t stream) {
  const float* x     = (const float*)d_in[0];
  const int*   batch = (const int*)d_in[1];
  const float* gfeat = (const float*)d_in[2];
  const float* W1    = (const float*)d_in[3];
  const float* b1    = (const float*)d_in[4];
  const float* W2    = (const float*)d_in[5];
  const float* b2    = (const float*)d_in[6];
  float*       out   = (float*)d_out;

  const int n    = in_sizes[1];          // N rows
  const int nseg = in_sizes[2] / GDIM;   // B segments

  float* s = (float*)d_ws;               // scratch: scores [n]

  const int tiles  = (n + 15) >> 4;
  int blocks = (tiles + 7) / 8;
  if (blocks > 2048) blocks = 2048;

  mlp_score_kernel<<<blocks, 256, 0, stream>>>(x, batch, gfeat, W1, b1, W2,
                                               b2, s, n);
  segment_softmax_kernel<<<(nseg + 255) / 256, 256, 0, stream>>>(s, batch, out,
                                                                 n, nseg);
}
